// GCAModel_26800595927373
// MI455X (gfx1250) — matrix operand order repaired and checked by reading, not compiled, not run on hardware
//
#include <hip/hip_runtime.h>

typedef __bf16 bf16;
typedef __attribute__((ext_vector_type(16))) __bf16 v16bf;
typedef __attribute__((ext_vector_type(8)))  float  v8f;

// ---- model dims ----
#define B_    4
#define CDD_  5
#define HIS_  50
#define L_    20
#define T_    41
#define E_    300
#define EP    320    // E padded to K-multiple (32)
#define H_    16
#define R_    256
#define Q_    200
#define NITEM 220    // B*CDD + B*HIS news items
#define NITR  1000   // B*CDD*HIS interaction items
#define INV_SCALE 0.05773502691896258f   // 1/sqrt(300) -- used in BOTH stages per reference

#define LDS_ALIGN __attribute__((aligned(32)))

// ---- WMMA helpers ----
__device__ __forceinline__ v8f wmma_bf16(v16bf a, v16bf b, v8f c) {
  return __builtin_amdgcn_wmma_f32_16x16x32_bf16(false, a, false, b, (short)0, c, false, false);
}
// A fragment from a row-major bf16 matrix: p = &mat[row][kt*32]
__device__ __forceinline__ v16bf afrag(const bf16* p, int abase) {
  v16bf a;
#pragma unroll
  for (int i = 0; i < 16; ++i) a[i] = p[abase + i];
  return a;
}
// B fragment, strided rows: p = &mat[kbase][n], rowStride in elements
__device__ __forceinline__ v16bf bfrag_strided(const bf16* p, int rowStride) {
  v16bf b;
#pragma unroll
  for (int i = 0; i < 16; ++i) b[i] = p[i * rowStride];
  return b;
}

// ---- CDNA5 async global->LDS copy (ASYNCcnt-tracked, no VGPR data) ----
// ISA 15.18.3 op 98: GLOBAL_LOAD_ASYNC_TO_LDS_B128  vdst(lds addr), vaddr(64b), off
// Generic shared pointers: addr[31:0] IS the LDS byte offset (ISA 10.2 aperture rules).
__device__ __forceinline__ void async_g2l_b128(unsigned int lds_addr, const void* gaddr) {
  asm volatile("global_load_async_to_lds_b128 %0, %1, off"
               :: "v"(lds_addr), "v"((unsigned long long)(uintptr_t)gaddr)
               : "memory");
}
__device__ __forceinline__ void wait_async0() {
  asm volatile("s_wait_asynccnt 0x0" ::: "memory");
}
__device__ __forceinline__ unsigned int lds_off(const void* p) {
  return (unsigned int)(uintptr_t)p;
}

// ============================================================================
// K1a: gather embeddings, pad [20,300] -> [32,320], convert to bf16
// ============================================================================
__global__ __launch_bounds__(256) void k_embed(const int* __restrict__ cdd,
                                               const int* __restrict__ his,
                                               const float* __restrict__ emb,
                                               bf16* __restrict__ xw) {
  int item = blockIdx.x;                       // 0..219
  bf16* dst = xw + (size_t)item * 32 * EP;
  for (int idx = threadIdx.x; idx < 32 * EP; idx += 256) {
    int row = idx / EP, col = idx % EP;
    float v = 0.f;
    if (row < L_ && col < E_) {
      int tok = (item < B_ * CDD_) ? cdd[item * L_ + row]
                                   : his[(item - B_ * CDD_) * L_ + row];
      v = emb[(size_t)tok * E_ + col];
    }
    dst[idx] = (bf16)v;
  }
}

// ============================================================================
// K1b: pack an fp32 [R,C] weight (zero-padded) into WMMA bf16 B-fragments:
//   dst[((batch*KT+kt)*NT+nt)*512 + lane*16 + i] = B[kt*32+(lane<16?0:16)+i][nt*16+lane%16]
// ============================================================================
__global__ void k_pack_bfrag(const float* __restrict__ src, int srcR, int srcC,
                             long srcBatchStride, int KT, int NT,
                             bf16* __restrict__ dst) {
  int frag  = blockIdx.x;                      // kt*NT + nt
  int batch = blockIdx.y;
  int kt = frag / NT, nt = frag % NT;
  int lane = threadIdx.x;                      // 0..31 (wave32)
  const float* s = src + (size_t)batch * srcBatchStride;
  bf16* d = dst + (((size_t)batch * KT + kt) * NT + nt) * 512 + lane * 16;
  int kb = kt * 32 + ((lane < 16) ? 0 : 16);
  int n = nt * 16 + (lane & 15);
#pragma unroll
  for (int i = 0; i < 16; ++i) {
    int k = kb + i;
    float v = (k < srcR && n < srcC) ? s[(size_t)k * srcC + n] : 0.f;
    d[i] = (bf16)v;
  }
}

// ============================================================================
// K2: word-level MHSA per news item. All three GEMMs (q-proj, scores, attn-out)
// on the matrix pipe; VALU does only softmax. x staged via async-to-LDS.
// ============================================================================
__global__ __launch_bounds__(256) void k_word_mhsa(const bf16* __restrict__ xw,
                                                   const bf16* __restrict__ wq,  // [16][10][19][512]
                                                   const bf16* __restrict__ wv,  // [16][10][1][512]
                                                   bf16* __restrict__ rep) {     // [220][20][256]
  __shared__ LDS_ALIGN bf16  x_l[32][EP];
  __shared__ LDS_ALIGN bf16  q_l[32][EP];     // bf16 q, cols 304..319 stay zero
  __shared__ LDS_ALIGN bf16  xv_l[32][16];
  __shared__ LDS_ALIGN float sc[32][33];
  __shared__ LDS_ALIGN bf16  a_l[32][32];     // softmaxed weights, zero-padded

  int item = blockIdx.x;
  int tid = threadIdx.x, wave = tid >> 5, lane = tid & 31;

  // async copy x (20480B = 1280 x b128 chunks) directly into LDS
  {
    unsigned int xl = lds_off(&x_l[0][0]);
    const char* xg = (const char*)(xw + (size_t)item * 32 * EP);
    for (int i = tid; i < (32 * EP * 2) / 16; i += 256)
      async_g2l_b128(xl + i * 16, xg + i * 16);
  }
  for (int i = tid; i < 32 * EP; i += 256) ((bf16*)q_l)[i] = (bf16)0.0f;
  for (int i = tid; i < 32 * 32; i += 256) ((bf16*)a_l)[i] = (bf16)0.0f;
  wait_async0();
  __syncthreads();

  int m     = lane & 15;
  int abase = (lane < 16) ? 0 : 16;   // A-fragment K base
  int kbase = (lane < 16) ? 0 : 16;   // B-fragment K base
  int mrow  = (lane < 16) ? 0 : 8;    // C-fragment row base

  for (int h = 0; h < H_; ++h) {
    const bf16* wqh = wq + (size_t)h * 10 * 19 * 512;
    const bf16* wvh = wv + (size_t)h * 10 * 512;
    if (h + 1 < H_) __builtin_prefetch(wq + (size_t)(h + 1) * 10 * 19 * 512 + lane * 64, 0, 1);

    // ---- q = x@Wq (2x19 tiles), K=320; constant 19*512 kt-stride ----
    for (int tile = wave; tile < 38; tile += 8) {
      int mt = tile / 19, nt = tile % 19;
      const bf16* wb = wqh + nt * 512 + lane * 16;
      v8f acc = {};
      for (int kt = 0; kt < 10; ++kt) {
        v16bf a = afrag(&x_l[mt * 16 + m][kt * 32], abase);
        v16bf b = *(const v16bf*)(wb + kt * (19 * 512));
        acc = wmma_bf16(a, b, acc);
      }
#pragma unroll
      for (int r = 0; r < 8; ++r)
        q_l[mt * 16 + mrow + r][nt * 16 + m] = (bf16)acc[r];
    }
    // ---- xv = x@Wv (2 tiles), constant 512 kt-stride ----
    for (int tile = wave; tile < 2; tile += 8) {
      const bf16* wb = wvh + lane * 16;
      v8f acc = {};
      for (int kt = 0; kt < 10; ++kt) {
        v16bf a = afrag(&x_l[tile * 16 + m][kt * 32], abase);
        v16bf b = *(const v16bf*)(wb + kt * 512);
        acc = wmma_bf16(a, b, acc);
      }
#pragma unroll
      for (int r = 0; r < 8; ++r)
        xv_l[tile * 16 + mrow + r][m] = (bf16)acc[r];
    }
    __syncthreads();

    // ---- scores = q @ x^T / sqrt(E): B frag is contiguous row of x ----
    for (int tile = wave; tile < 4; tile += 8) {
      int mt = tile >> 1, nt = tile & 1;
      v8f acc = {};
      for (int kt = 0; kt < 10; ++kt) {
        v16bf a = afrag(&q_l[mt * 16 + m][kt * 32], abase);
        v16bf b = *(const v16bf*)&x_l[nt * 16 + m][kt * 32 + kbase];
        acc = wmma_bf16(a, b, acc);
      }
#pragma unroll
      for (int r = 0; r < 8; ++r)
        sc[mt * 16 + mrow + r][nt * 16 + m] = acc[r] * INV_SCALE;
    }
    __syncthreads();

    // ---- softmax rows -> bf16 attention weights ----
    if (tid < L_) {
      float mx = -1e30f;
      for (int s = 0; s < L_; ++s) mx = fmaxf(mx, sc[tid][s]);
      float sum = 0.f;
      float e[L_];
      for (int s = 0; s < L_; ++s) { e[s] = __expf(sc[tid][s] - mx); sum += e[s]; }
      float inv = 1.f / sum;
      for (int s = 0; s < L_; ++s) a_l[tid][s] = (bf16)(e[s] * inv);
    }
    __syncthreads();

    // ---- out = a @ xv (2 tiles, K=32) -> rep[item][t][h*16+j] ----
    for (int tile = wave; tile < 2; tile += 8) {
      v8f acc = {};
      v16bf a = afrag(&a_l[tile * 16 + m][0], abase);
      v16bf b = bfrag_strided(&xv_l[kbase][m], 16);
      acc = wmma_bf16(a, b, acc);
#pragma unroll
      for (int r = 0; r < 8; ++r) {
        int row = tile * 16 + mrow + r;
        if (row < L_) rep[((size_t)item * L_ + row) * R_ + h * 16 + m] = (bf16)acc[r];
      }
    }
    __syncthreads();
  }
}

// ============================================================================
// K3: interaction transformer + additive pooling per (b,c,his) item.
// fused:[48,256]bf16 (41 real rows: cdd(20) | ones | his(20)); all GEMMs WMMA.
// cdd/his blocks staged via async-to-LDS.
// ============================================================================
__global__ __launch_bounds__(256) void k_itr(const bf16* __restrict__ rep,   // [220][20][256]
                                             const bf16* __restrict__ wq,    // [16][8][16][512]
                                             const bf16* __restrict__ wv,    // [16][8][1][512]
                                             const bf16* __restrict__ kw,    // [8][13][512]
                                             const float* __restrict__ keyb, // [200]
                                             const float* __restrict__ query,// [200]
                                             float* __restrict__ rep_itr) {  // [1000][256]
  __shared__ LDS_ALIGN bf16  f_l[48][R_];
  __shared__ LDS_ALIGN bf16  q_l[48][R_];     // q (bf16); later reused for tanh(key) result
  __shared__ LDS_ALIGN bf16  val_l[48][R_];
  __shared__ LDS_ALIGN bf16  fv_l[64][16];    // rows 48..63 zero
  __shared__ LDS_ALIGN float sc[48][49];
  __shared__ LDS_ALIGN bf16  a_l[48][64];     // softmaxed weights, zero-padded to K=64
  __shared__ float att[T_];

  int tid = threadIdx.x, wave = tid >> 5, lane = tid & 31;
  int hi = blockIdx.x % HIS_;
  int bc = blockIdx.x / HIS_;       // b*CDD + c
  int b  = bc / CDD_;
  int item_c = bc;
  int item_h = B_ * CDD_ + b * HIS_ + hi;

  // async copy cdd block -> rows 0..19, his block -> rows 21..40 (10240B each)
  {
    unsigned int fl = lds_off(&f_l[0][0]);
    const char* gc = (const char*)(rep + (size_t)item_c * 20 * R_);
    const char* gh = (const char*)(rep + (size_t)item_h * 20 * R_);
    for (int i = tid; i < (20 * R_ * 2) / 16; i += 256) {          // 640 chunks
      async_g2l_b128(fl + i * 16, gc + i * 16);
      async_g2l_b128(fl + 21 * R_ * 2 + i * 16, gh + i * 16);
    }
  }
  for (int i = tid; i < R_; i += 256) f_l[20][i] = (bf16)1.0f;     // separator row
  for (int i = tid; i < 7 * R_; i += 256) f_l[41 + i / R_][i % R_] = (bf16)0.0f;
  for (int i = tid; i < 48 * 64; i += 256) ((bf16*)a_l)[i] = (bf16)0.0f;
  for (int i = tid; i < 16 * 16; i += 256) fv_l[48 + i / 16][i % 16] = (bf16)0.0f;
  wait_async0();
  __syncthreads();

  int m     = lane & 15;
  int abase = (lane < 16) ? 0 : 16;
  int kbase = (lane < 16) ? 0 : 16;
  int mrow  = (lane < 16) ? 0 : 8;

  for (int h = 0; h < H_; ++h) {
    const bf16* wqh = wq + (size_t)h * 8 * 16 * 512;
    const bf16* wvh = wv + (size_t)h * 8 * 512;
    if (h + 1 < H_) __builtin_prefetch(wq + (size_t)(h + 1) * 8 * 16 * 512 + lane * 64, 0, 1);

    // ---- q = fused@Wq (3x16 tiles), K=256; constant 16*512 kt-stride ----
    for (int tile = wave; tile < 48; tile += 8) {
      int mt = tile / 16, nt = tile % 16;
      const bf16* wb = wqh + nt * 512 + lane * 16;
      v8f acc = {};
      for (int kt = 0; kt < 8; ++kt) {
        v16bf a = afrag(&f_l[mt * 16 + m][kt * 32], abase);
        v16bf bfr = *(const v16bf*)(wb + kt * (16 * 512));
        acc = wmma_bf16(a, bfr, acc);
      }
#pragma unroll
      for (int r = 0; r < 8; ++r)
        q_l[mt * 16 + mrow + r][nt * 16 + m] = (bf16)acc[r];
    }
    // ---- fv = fused@Wv (3 tiles), constant 512 kt-stride ----
    for (int tile = wave; tile < 3; tile += 8) {
      const bf16* wb = wvh + lane * 16;
      v8f acc = {};
      for (int kt = 0; kt < 8; ++kt) {
        v16bf a = afrag(&f_l[tile * 16 + m][kt * 32], abase);
        v16bf bfr = *(const v16bf*)(wb + kt * 512);
        acc = wmma_bf16(a, bfr, acc);
      }
#pragma unroll
      for (int r = 0; r < 8; ++r)
        fv_l[tile * 16 + mrow + r][m] = (bf16)acc[r];
    }
    __syncthreads();

    // ---- scores = q @ fused^T / sqrt(E): 3x3 tiles, K=256 ----
    for (int tile = wave; tile < 9; tile += 8) {
      int mt = tile / 3, nt = tile % 3;
      v8f acc = {};
      for (int kt = 0; kt < 8; ++kt) {
        v16bf a = afrag(&q_l[mt * 16 + m][kt * 32], abase);
        v16bf bfr = *(const v16bf*)&f_l[nt * 16 + m][kt * 32 + kbase];
        acc = wmma_bf16(a, bfr, acc);
      }
#pragma unroll
      for (int r = 0; r < 8; ++r)
        sc[mt * 16 + mrow + r][nt * 16 + m] = acc[r] * INV_SCALE;
    }
    __syncthreads();

    // ---- softmax rows -> bf16 weights ----
    if (tid < T_) {
      float mx = -1e30f;
      for (int s = 0; s < T_; ++s) mx = fmaxf(mx, sc[tid][s]);
      float sum = 0.f;
      float e[T_];
      for (int s = 0; s < T_; ++s) { e[s] = __expf(sc[tid][s] - mx); sum += e[s]; }
      float inv = 1.f / sum;
      for (int s = 0; s < T_; ++s) a_l[tid][s] = (bf16)(e[s] * inv);
    }
    __syncthreads();

    // ---- out = a @ fv (3 tiles, K=64) -> val (pad rows zeroed for free) ----
    for (int tile = wave; tile < 3; tile += 8) {
      v8f acc = {};
      for (int kt = 0; kt < 2; ++kt) {
        v16bf a = afrag(&a_l[tile * 16 + m][kt * 32], abase);
        v16bf bfr = bfrag_strided(&fv_l[kt * 32 + kbase][m], 16);
        acc = wmma_bf16(a, bfr, acc);
      }
#pragma unroll
      for (int r = 0; r < 8; ++r)
        val_l[tile * 16 + mrow + r][h * 16 + m] = (bf16)acc[r];
    }
    __syncthreads();
  }

  // ---- key projection: k = tanh(val @ keyW + keyb), bias in C-init ----
  for (int tile = wave; tile < 39; tile += 8) {   // 3 x 13 tiles
    int mt = tile / 13, nt = tile % 13;
    int col = nt * 16 + m;
    float bias = (col < Q_) ? keyb[col] : 0.f;
    const bf16* wb = kw + nt * 512 + lane * 16;
    v8f acc;
#pragma unroll
    for (int r = 0; r < 8; ++r) acc[r] = bias;
    for (int kt = 0; kt < 8; ++kt) {
      v16bf a = afrag(&val_l[mt * 16 + m][kt * 32], abase);
      v16bf bfr = *(const v16bf*)(wb + kt * (13 * 512));
      acc = wmma_bf16(a, bfr, acc);
    }
#pragma unroll
    for (int r = 0; r < 8; ++r)
      q_l[mt * 16 + mrow + r][nt * 16 + m] = (bf16)tanhf(acc[r]);
  }
  __syncthreads();

  // ---- additive attention logits + softmax + pooled representation ----
  if (tid < T_) {
    float acc = 0.f;
    for (int k = 0; k < Q_; ++k) acc += query[k] * (float)q_l[tid][k];
    att[tid] = acc * INV_SCALE;
  }
  __syncthreads();
  if (tid == 0) {
    float mx = -1e30f;
    for (int t = 0; t < T_; ++t) mx = fmaxf(mx, att[t]);
    float sum = 0.f;
    for (int t = 0; t < T_; ++t) { float e = __expf(att[t] - mx); att[t] = e; sum += e; }
    float inv = 1.f / sum;
    for (int t = 0; t < T_; ++t) att[t] *= inv;
  }
  __syncthreads();
  if (tid < R_) {
    float acc = 0.f;
    for (int t = 0; t < T_; ++t) acc += att[t] * (float)val_l[t][tid];
    rep_itr[(size_t)blockIdx.x * R_ + tid] = acc;
  }
}

// ============================================================================
// K4: mean over HIS, LTR projection, log_softmax over CDD
// ============================================================================
__global__ __launch_bounds__(256) void k_final(const float* __restrict__ rep_itr,
                                               const float* __restrict__ ltr_w,
                                               const float* __restrict__ ltr_b,
                                               float* __restrict__ out) {
  __shared__ float red[256];
  __shared__ float score[B_ * CDD_];
  int tid = threadIdx.x;
  for (int p = 0; p < B_ * CDD_; ++p) {
    float s = 0.f;
    for (int hh = 0; hh < HIS_; ++hh)
      s += rep_itr[((size_t)p * HIS_ + hh) * R_ + tid];
    red[tid] = (s / (float)HIS_) * ltr_w[tid];
    __syncthreads();
    for (int off = 128; off > 0; off >>= 1) {
      if (tid < off) red[tid] += red[tid + off];
      __syncthreads();
    }
    if (tid == 0) score[p] = red[0] + ltr_b[0];
    __syncthreads();
  }
  if (tid < B_) {
    float mx = -1e30f;
    for (int c = 0; c < CDD_; ++c) mx = fmaxf(mx, score[tid * CDD_ + c]);
    float sum = 0.f;
    for (int c = 0; c < CDD_; ++c) sum += __expf(score[tid * CDD_ + c] - mx);
    float lse = mx + __logf(sum);
    for (int c = 0; c < CDD_; ++c) out[tid * CDD_ + c] = score[tid * CDD_ + c] - lse;
  }
}

// ============================================================================
extern "C" void kernel_launch(void* const* d_in, const int* in_sizes, int n_in,
                              void* d_out, int out_size, void* d_ws, size_t ws_size,
                              hipStream_t stream) {
  (void)in_sizes; (void)n_in; (void)out_size; (void)ws_size;
  const int*   cdd   = (const int*)d_in[0];
  const int*   his   = (const int*)d_in[1];
  const float* emb   = (const float*)d_in[2];
  const float* Wq_w  = (const float*)d_in[3];
  const float* Wv_w  = (const float*)d_in[4];
  const float* Wq_i  = (const float*)d_in[5];
  const float* Wv_i  = (const float*)d_in[6];
  const float* keyW  = (const float*)d_in[7];
  const float* keyb  = (const float*)d_in[8];
  const float* query = (const float*)d_in[9];
  const float* ltrw  = (const float*)d_in[10];
  const float* ltrb  = (const float*)d_in[11];

  char* ws = (char*)d_ws;
  size_t off = 0;
  auto alloc = [&](size_t bytes) -> void* {
    void* p = ws + off;
    off = (off + bytes + 255) & ~(size_t)255;
    return p;
  };
  bf16*  xw       = (bf16*)alloc((size_t)NITEM * 32 * EP * 2);      // padded embeddings
  bf16*  rep_word = (bf16*)alloc((size_t)NITEM * 20 * R_ * 2);      // word-MHSA output
  bf16*  wq_w_p   = (bf16*)alloc((size_t)16 * 10 * 19 * 512 * 2);   // packed Wq_words
  bf16*  wv_w_p   = (bf16*)alloc((size_t)16 * 10 * 1  * 512 * 2);   // packed Wv_words
  bf16*  wq_i_p   = (bf16*)alloc((size_t)16 * 8  * 16 * 512 * 2);   // packed Wq_itrs
  bf16*  wv_i_p   = (bf16*)alloc((size_t)16 * 8  * 1  * 512 * 2);   // packed Wv_itrs
  bf16*  keyW_p   = (bf16*)alloc((size_t)8  * 13 * 512 * 2);        // packed keyW
  float* rep_itr  = (float*)alloc((size_t)NITR * R_ * 4);           // pooled interaction reprs

  k_embed<<<NITEM, 256, 0, stream>>>(cdd, his, emb, xw);
  k_pack_bfrag<<<dim3(10 * 19, 16), 32, 0, stream>>>(Wq_w, 300, 300, 90000, 10, 19, wq_w_p);
  k_pack_bfrag<<<dim3(10 * 1,  16), 32, 0, stream>>>(Wv_w, 300, 16,  4800,  10, 1,  wv_w_p);
  k_pack_bfrag<<<dim3(8 * 16,  16), 32, 0, stream>>>(Wq_i, 256, 256, 65536, 8,  16, wq_i_p);
  k_pack_bfrag<<<dim3(8 * 1,   16), 32, 0, stream>>>(Wv_i, 256, 16,  4096,  8,  1,  wv_i_p);
  k_pack_bfrag<<<dim3(8 * 13,  1),  32, 0, stream>>>(keyW, 256, 200, 0,     8,  13, keyW_p);
  k_word_mhsa<<<NITEM, 256, 0, stream>>>(xw, wq_w_p, wv_w_p, rep_word);
  k_itr<<<NITR, 256, 0, stream>>>(rep_word, wq_i_p, wv_i_p, keyW_p, keyb, query, rep_itr);
  k_final<<<1, 256, 0, stream>>>(rep_itr, ltrw, ltrb, (float*)d_out);
}
